// BayesianLSTM_41145786696005
// MI455X (gfx1250) — compile-verified, matrix operand order checked by reading, not run on hardware
//
#include <hip/hip_runtime.h>
#include <hip/hip_bf16.h>
#include <cstdint>

// ---------------------------------------------------------------------------
// BayesianLSTM forward for MI455X (gfx1250, wave32, WMMA).
//   - bf16 weights (converted once into workspace), f32 accumulate via
//     v_wmma_f32_16x16x32_bf16.
//   - Per-timestep fused kernel: GEMM (M=256,N=4096,K=1024) + LSTM gates.
//   - 4 waves per workgroup, one gate per wave.
//   - Activation operand staged via ASYNC global->LDS DMA, double-buffered
//     (ping-pong) so the copy overlaps the WMMA chain; one s_wait_asynccnt
//     + one workgroup barrier per 64-wide K chunk.
// ---------------------------------------------------------------------------

typedef __attribute__((ext_vector_type(16))) __bf16          v16bf;
typedef __attribute__((ext_vector_type(8)))  float           v8f;
typedef __attribute__((ext_vector_type(8)))  unsigned short  v8us;

constexpr int kH  = 1024;   // hidden size
constexpr int kB  = 256;    // batch
constexpr int kNG = 4096;   // 4*kH gate columns
constexpr int kTEnc = 84;   // encoder steps (168 conv -> pool/2)
constexpr int kTOut = 24;   // decoder outputs
constexpr int kKC = 64;     // K elements staged per async chunk

__device__ __forceinline__ unsigned short f2bf(float f) {
  unsigned int u = __float_as_uint(f);
  unsigned int r = (u + 0x7FFFu + ((u >> 16) & 1u)) >> 16;  // RNE
  return (unsigned short)r;
}
__device__ __forceinline__ float sigm(float x) { return 1.f / (1.f + __expf(-x)); }

__device__ __forceinline__ unsigned int mix_hash(unsigned int t, unsigned int b, unsigned int k) {
  unsigned int h = t * 0x9E3779B1u ^ (b + 0x85EBCA6Bu) * 0xC2B2AE35u ^ (k + 0x27D4EB2Fu) * 0x165667B1u;
  h ^= h >> 15; h *= 0x2C1B3C6Du; h ^= h >> 12; h *= 0x297A2D39u; h ^= h >> 15;
  return h;
}

// Async DMA: 16 bytes global -> LDS, tracked by ASYNCcnt (no VGPR round-trip).
__device__ __forceinline__ void async_g2l_b128(unsigned lds_byte_off, const void* gaddr) {
  asm volatile("global_load_async_to_lds_b128 %0, %1, off"
               :: "v"(lds_byte_off), "v"(gaddr) : "memory");
}
__device__ __forceinline__ void wait_async0() {
#if __has_builtin(__builtin_amdgcn_s_wait_asynccnt)
  __builtin_amdgcn_s_wait_asynccnt(0);
#else
  asm volatile("s_wait_asynccnt 0x0" ::: "memory");
#endif
}

// ---- one-time: fp32 -> bf16 weights + fused biases -------------------------
__global__ void prep_kernel(const float* __restrict__ Whh0, const float* __restrict__ Wih1,
                            const float* __restrict__ Whh1,
                            const float* __restrict__ bih0, const float* __restrict__ bhh0,
                            const float* __restrict__ bih1, const float* __restrict__ bhh1,
                            unsigned short* __restrict__ Whh0b, unsigned short* __restrict__ Wih1b,
                            unsigned short* __restrict__ Whh1b,
                            float* __restrict__ b0, float* __restrict__ b1) {
  int idx = blockIdx.x * blockDim.x + threadIdx.x;
  if (idx < kNG * kH) {
    Whh0b[idx] = f2bf(Whh0[idx]);
    Wih1b[idx] = f2bf(Wih1[idx]);
    Whh1b[idx] = f2bf(Whh1[idx]);
  }
  if (idx < kNG) { b0[idx] = bih0[idx] + bhh0[idx]; b1[idx] = bih1[idx] + bhh1[idx]; }
}

// ---- one-time: zero all recurrent state ------------------------------------
__global__ void init_kernel(float* c0, float* c1, float* h0f, float* h1f,
                            unsigned short* h0b0, unsigned short* h0b1,
                            unsigned short* h1b0, unsigned short* h1b1, float* yprev) {
  int idx = blockIdx.x * blockDim.x + threadIdx.x;
  if (idx < kB * kH) {
    c0[idx] = 0.f; c1[idx] = 0.f; h0f[idx] = 0.f; h1f[idx] = 0.f;
    h0b0[idx] = 0; h0b1[idx] = 0; h1b0[idx] = 0; h1b1[idx] = 0;
  }
  if (idx < kB) yprev[idx] = 0.f;
}

// ---- conv1d(pad=1) + relu + maxpool/2 -> seq[84][256] ----------------------
__global__ void conv_pool_kernel(const float* __restrict__ x, const float* __restrict__ cw,
                                 const float* __restrict__ cb, float* __restrict__ seq) {
  int idx = blockIdx.x * blockDim.x + threadIdx.x;
  if (idx >= kTEnc * kB) return;
  int b = idx & (kB - 1);
  int t84 = idx >> 8;
  float best = -3.4e38f;
  #pragma unroll
  for (int p = 0; p < 2; ++p) {
    int t = t84 * 2 + p;
    float s = cb[0];
    #pragma unroll
    for (int kk = 0; kk < 3; ++kk) {
      int tau = t + kk - 1;
      if (tau < 0 || tau >= 168) continue;
      const float* xr = x + ((size_t)b * 192 + tau) * 8;
      #pragma unroll
      for (int f = 0; f < 8; ++f) s += xr[f] * cw[f * 3 + kk];
    }
    s = fmaxf(s, 0.f);
    best = fmaxf(best, s);
  }
  seq[(size_t)t84 * kB + b] = best;
}

// ---- fused LSTM layer step: D = sum_pairs A@W^T (+ bias + x*wih0), gates ----
// grid (8,16): block tile 32 batch rows x 64 hidden cols (x4 gates).
// 128 threads = 4 waves; wave w computes gate w's 32x64 tile via 2x4 WMMA tiles.
__global__ __launch_bounds__(128) void lstm_step_kernel(
    const unsigned short* __restrict__ A1, const unsigned short* __restrict__ W1,
    const unsigned short* __restrict__ A2, const unsigned short* __restrict__ W2,
    int npairs,
    const float* __restrict__ bias,
    const float* __restrict__ xv, const float* __restrict__ w0col,
    float* __restrict__ c_state, float* __restrict__ h_out_f,
    unsigned short* __restrict__ h_out_b) {
  __shared__ __align__(32) unsigned short As[2][32 * kKC];  // ping-pong, 4KB each
  __shared__ float Gx[4][32 * 16];                          // gate exchange slabs

  const int tid  = threadIdx.x;
  const int lane = tid & 31;
  const int wv   = tid >> 5;            // gate index 0..3
  const int m0   = blockIdx.x * 32;     // batch rows
  const int j0   = blockIdx.y * 64;     // hidden columns
  const int hl   = lane >> 4;           // which K-half this lane carries
  const int ln   = lane & 15;

  v8f acc[2][4];
  #pragma unroll
  for (int mi = 0; mi < 2; ++mi)
    #pragma unroll
    for (int ji = 0; ji < 4; ++ji)
      #pragma unroll
      for (int e = 0; e < 8; ++e) acc[mi][ji][e] = 0.f;

  const int chunks_per_pair = kH / kKC;              // 16
  const int total_chunks = npairs * chunks_per_pair; // 16 or 32

  // Stage chunk ch's 32 x 64 bf16 A-tile into As[buf] via async DMA.
  auto stage = [&](int buf, int ch) {
    const unsigned short* Ab = (ch >= chunks_per_pair) ? A2 : A1;
    int k0 = (ch & (chunks_per_pair - 1)) * kKC;
    #pragma unroll
    for (int c = 0; c < 2; ++c) {                    // 128 thr x 2 x 16B = 4KB
      int u   = tid * 8 + c * 1024;                  // ushort index 0..2047
      int row = u >> 6;
      int kk  = u & 63;
      unsigned lds = (unsigned)(unsigned long long)(const void*)&As[buf][row * kKC + kk];
      async_g2l_b128(lds, &Ab[(size_t)(m0 + row) * kH + k0 + kk]);
    }
  };

  stage(0, 0);                                       // prologue prefetch
  int buf = 0;
  for (int ch = 0; ch < total_chunks; ++ch) {
    wait_async0();                                   // As[buf] DMA complete
    __syncthreads();                                 // visible to all waves
    if (ch + 1 < total_chunks) stage(buf ^ 1, ch + 1);  // overlap next DMA

    const unsigned short* Wb = (ch >= chunks_per_pair) ? W2 : W1;
    int k0 = (ch & (chunks_per_pair - 1)) * kKC;
    #pragma unroll
    for (int ks = 0; ks < kKC; ks += 32) {
      v16bf afrag[2];
      #pragma unroll
      for (int mi = 0; mi < 2; ++mi)
        afrag[mi] = *(const v16bf*)&As[buf][(mi * 16 + ln) * kKC + ks + hl * 16];
      #pragma unroll
      for (int ji = 0; ji < 4; ++ji) {
        int n = wv * kH + j0 + ji * 16 + ln;  // gate column in W (row-major over K)
        v16bf bfrag = *(const v16bf*)&Wb[(size_t)n * kH + k0 + ks + hl * 16];
        #pragma unroll
        for (int mi = 0; mi < 2; ++mi)
          acc[mi][ji] = __builtin_amdgcn_wmma_f32_16x16x32_bf16(
              false, afrag[mi], false, bfrag, (short)0, acc[mi][ji], false, false);
      }
    }
    buf ^= 1;
  }

  // Epilogue: exchange gates through LDS one 16-column slab at a time.
  for (int ji = 0; ji < 4; ++ji) {
    __syncthreads();
    #pragma unroll
    for (int mi = 0; mi < 2; ++mi)
      #pragma unroll
      for (int e = 0; e < 8; ++e) {
        int ml = mi * 16 + e + ((lane & 16) ? 8 : 0);  // C-layout: vgpr e -> M
        Gx[wv][ml * 16 + ln] = acc[mi][ji][e];
      }
    __syncthreads();
    #pragma unroll
    for (int rep = 0; rep < 4; ++rep) {
      int e  = rep * 128 + tid;      // 0..511 over 32x16 slab
      int ml = e >> 4;
      int jj = e & 15;
      int j  = j0 + ji * 16 + jj;
      int m  = m0 + ml;
      float riv = Gx[0][ml * 16 + jj] + bias[j];
      float rfv = Gx[1][ml * 16 + jj] + bias[kH + j];
      float rgv = Gx[2][ml * 16 + jj] + bias[2 * kH + j];
      float rov = Gx[3][ml * 16 + jj] + bias[3 * kH + j];
      if (xv) {                       // layer-0 rank-1 input term (input dim 1)
        float xb = xv[m];
        riv += xb * w0col[j];
        rfv += xb * w0col[kH + j];
        rgv += xb * w0col[2 * kH + j];
        rov += xb * w0col[3 * kH + j];
      }
      size_t off = (size_t)m * kH + j;
      float cold = c_state[off];
      float cn = sigm(rfv) * cold + sigm(riv) * tanhf(rgv);
      float hn = sigm(rov) * tanhf(cn);
      c_state[off] = cn;
      h_out_f[off] = hn;
      h_out_b[off] = f2bf(hn);
    }
  }
}

// ---- FC head with hash-Bernoulli(0.8) dropout; one block per batch row -----
__global__ __launch_bounds__(256) void fc_kernel(const float* __restrict__ h1,
                                                 const float* __restrict__ fcw,
                                                 const float* __restrict__ fcb,
                                                 int t, float* __restrict__ out,
                                                 float* __restrict__ yprev) {
  __shared__ float red[256];
  int b = blockIdx.x;
  int tid = threadIdx.x;
  float s = 0.f;
  #pragma unroll
  for (int r = 0; r < 4; ++r) {
    int k = tid + r * 256;
    unsigned int hsh = mix_hash((unsigned)t, (unsigned)b, (unsigned)k);
    bool keep = (hsh & 0xFFFFFFu) < 13421773u;  // 0.8 * 2^24
    if (keep) s += h1[(size_t)b * kH + k] * fcw[k] * 1.25f;  // 1/KEEP
  }
  red[tid] = s;
  __syncthreads();
  for (int stp = 128; stp > 0; stp >>= 1) {
    if (tid < stp) red[tid] += red[tid + stp];
    __syncthreads();
  }
  if (tid == 0) {
    float y = red[0] + fcb[0];
    out[(size_t)b * kTOut + t] = y;
    yprev[b] = y;
  }
}

// ---------------------------------------------------------------------------
extern "C" void kernel_launch(void* const* d_in, const int* in_sizes, int n_in,
                              void* d_out, int out_size, void* d_ws, size_t ws_size,
                              hipStream_t stream) {
  const float* x     = (const float*)d_in[0];
  const float* convw = (const float*)d_in[1];
  const float* convb = (const float*)d_in[2];
  const float* Wih0  = (const float*)d_in[3];   // [4096,1] -> rank-1 column
  const float* Whh0  = (const float*)d_in[4];
  const float* bih0  = (const float*)d_in[5];
  const float* bhh0  = (const float*)d_in[6];
  const float* Wih1  = (const float*)d_in[7];
  const float* Whh1  = (const float*)d_in[8];
  const float* bih1  = (const float*)d_in[9];
  const float* bhh1  = (const float*)d_in[10];
  const float* fcw   = (const float*)d_in[11];
  const float* fcb   = (const float*)d_in[12];
  float* out = (float*)d_out;

  // Workspace layout (~31 MB), 256B aligned slices.
  char* ws = (char*)d_ws;
  size_t o = 0;
  auto alloc = [&](size_t bytes) -> char* {
    char* p = ws + o;
    o = (o + bytes + 255) & ~(size_t)255;
    return p;
  };
  unsigned short* Whh0b = (unsigned short*)alloc((size_t)kNG * kH * 2);
  unsigned short* Wih1b = (unsigned short*)alloc((size_t)kNG * kH * 2);
  unsigned short* Whh1b = (unsigned short*)alloc((size_t)kNG * kH * 2);
  float* b0   = (float*)alloc((size_t)kNG * 4);
  float* b1   = (float*)alloc((size_t)kNG * 4);
  float* c0   = (float*)alloc((size_t)kB * kH * 4);
  float* c1   = (float*)alloc((size_t)kB * kH * 4);
  float* h0f  = (float*)alloc((size_t)kB * kH * 4);
  float* h1f  = (float*)alloc((size_t)kB * kH * 4);
  unsigned short* h0b[2] = {(unsigned short*)alloc((size_t)kB * kH * 2),
                            (unsigned short*)alloc((size_t)kB * kH * 2)};
  unsigned short* h1b[2] = {(unsigned short*)alloc((size_t)kB * kH * 2),
                            (unsigned short*)alloc((size_t)kB * kH * 2)};
  float* seq   = (float*)alloc((size_t)kTEnc * kB * 4);
  float* yprev = (float*)alloc((size_t)kB * 4);
  (void)ws_size; (void)in_sizes; (void)n_in; (void)out_size;

  prep_kernel<<<(kNG * kH + 255) / 256, 256, 0, stream>>>(
      Whh0, Wih1, Whh1, bih0, bhh0, bih1, bhh1, Whh0b, Wih1b, Whh1b, b0, b1);
  init_kernel<<<(kB * kH + 255) / 256, 256, 0, stream>>>(
      c0, c1, h0f, h1f, h0b[0], h0b[1], h1b[0], h1b[1], yprev);
  conv_pool_kernel<<<(kTEnc * kB + 255) / 256, 256, 0, stream>>>(x, convw, convb, seq);

  dim3 grid(kB / 32, kH / 64);  // (8,16)
  const int nsteps = kTEnc + (kTOut - 1);  // 84 encoder + 23 decoder = 107
  for (int s = 0; s < nsteps; ++s) {
    const float* xvv = (s < kTEnc) ? (seq + (size_t)s * kB) : yprev;
    unsigned short* h0r = h0b[s & 1];
    unsigned short* h0w = h0b[(s + 1) & 1];
    unsigned short* h1r = h1b[s & 1];
    unsigned short* h1w = h1b[(s + 1) & 1];
    // layer 0: g = x*Wih0^T (rank-1) + h0@Whh0^T + b0
    lstm_step_kernel<<<grid, 128, 0, stream>>>(
        h0r, Whh0b, nullptr, nullptr, 1, b0, xvv, Wih0, c0, h0f, h0w);
    // layer 1: g = h0@Wih1^T + h1@Whh1^T + b1
    lstm_step_kernel<<<grid, 128, 0, stream>>>(
        h0w, Wih1b, h1r, Whh1b, 2, b1, nullptr, nullptr, c1, h1f, h1w);
    if (s >= kTEnc - 1)  // emit y_0 .. y_23
      fc_kernel<<<kB, 256, 0, stream>>>(h1f, fcw, fcb, s - (kTEnc - 1), out, yprev);
  }
}
